// MessagePassingLayer_31559419691865
// MI455X (gfx1250) — matrix operand order repaired and checked by reading, not compiled, not run on hardware
//
#include <hip/hip_runtime.h>
#include <hip/hip_bf16.h>

typedef __attribute__((ext_vector_type(16))) _Float16 v16h;
typedef __attribute__((ext_vector_type(4)))  _Float16 v4h;
typedef __attribute__((ext_vector_type(8)))  float    v8f;

#define DNODE 64
#define DIN   192   // 2*DNODE + DEDGE

// ---------------------------------------------------------------------------
// A-fragment loader: 16x32 f16 A tile for v_wmma_f32_16x16x32_f16.
// ISA layout (16-bit A 16x32): lanes 0-15 -> M=lane, VGPR0..3 = K 0..7 (pairs),
// VGPR4..7 = K 16..23; lanes 16-31 -> K 8..15 / 24..31.  Pairs (h, h+1) are
// K-contiguous so we read 2 halfs at a time from the row-major LDS buffer.
// ---------------------------------------------------------------------------
__device__ __forceinline__ v16h load_a16(const _Float16* __restrict__ base,
                                         int stride, int lane, int kc) {
  const int m  = lane & 15;
  const int kb = kc * 32 + ((lane & 16) ? 8 : 0);
  const _Float16* p = base + m * stride + kb;
  v16h a;
#pragma unroll
  for (int h = 0; h < 16; h += 2) {
    int k = (h & 7) + ((h & 8) ? 16 : 0);
    a[h]     = p[k];
    a[h + 1] = p[k + 1];
  }
  return a;
}

__device__ __forceinline__ float silu(float x) {
  return x / (1.0f + __expf(-x));
}

// ---------------------------------------------------------------------------
// Zero the aggregation buffer (must run every call: atomics accumulate).
// ---------------------------------------------------------------------------
__global__ void k_zero(float* __restrict__ p, int n) {
  int i = blockIdx.x * blockDim.x + threadIdx.x;
  if (i < n) p[i] = 0.0f;
}

// ---------------------------------------------------------------------------
// Pre-swizzle weights into per-lane WMMA B-fragment layout.
// B (32x16 f16): lanes 0-15 hold column N=lane, K=kc*32+0..7,16..23;
//                lanes 16-31 hold column N=lane-16, K=kc*32+8..15,24..31.
// Fragment f stored as 512 contiguous halfs: lane*16 + h.
// Frag order: Wm1 (6 kc x 4 nt = 24), Wm2 (8), Wu1 (8), Wu2 (8)  -> 48 frags.
// ---------------------------------------------------------------------------
__global__ void k_prep(const float* __restrict__ Wm1, const float* __restrict__ Wm2,
                       const float* __restrict__ Wu1, const float* __restrict__ Wu2,
                       _Float16* __restrict__ frag) {
  int idx = blockIdx.x * blockDim.x + threadIdx.x;
  if (idx >= 48 * 512) return;
  int fg = idx >> 9;          // fragment 0..47
  int r  = idx & 511;
  int lane = r >> 4, h = r & 15;
  const float* W; int fl;
  if (fg < 24)      { W = Wm1; fl = fg;      }
  else if (fg < 32) { W = Wm2; fl = fg - 24; }
  else if (fg < 40) { W = Wu1; fl = fg - 32; }
  else              { W = Wu2; fl = fg - 40; }
  int kc = fl >> 2, nt = fl & 3;
  int n = nt * 16 + (lane & 15);
  int k = kc * 32 + ((lane & 16) ? 8 : 0) + (h & 7) + ((h & 8) ? 16 : 0);
  frag[idx] = (_Float16)W[k * DNODE + n];
}

// ---------------------------------------------------------------------------
// Edge kernel: 64 edges per block (4 waves x 16-edge WMMA tiles).
//   layer1: [16,192]x[192,64]  -> 6x4 WMMA + SiLU
//   layer2: [16,64] x[64,64]   -> 2x4 WMMA
//   scatter: atomic f32 add into agg[edge_dst]
// ---------------------------------------------------------------------------
__global__ __launch_bounds__(128) void k_edge(
    const float* __restrict__ node_feat,
    const int*   __restrict__ edge_src,
    const int*   __restrict__ edge_dst,
    const float* __restrict__ edge_feat,
    const _Float16* __restrict__ wm1f,   // 24 frags
    const _Float16* __restrict__ wm2f,   // 8 frags
    const float* __restrict__ b_m1,
    const float* __restrict__ b_m2,
    float* __restrict__ agg,
    int E) {
  __shared__ _Float16 feat[64 * DIN];    // gathered msg_input, f16
  __shared__ _Float16 hbuf[64 * DNODE];  // layer-1 activations
  __shared__ int esrc[64];
  __shared__ int edst[64];

  const int tid = threadIdx.x;
  const int e0  = blockIdx.x * 64;

  if (tid < 64) {
    int e = e0 + tid;
    esrc[tid] = (e < E) ? edge_src[e] : -1;
    edst[tid] = (e < E) ? edge_dst[e] : -1;
  }
  __syncthreads();

  // gather: 48 float4 per edge -> f16 LDS rows [le][192]
  for (int i = tid; i < 64 * 48; i += 128) {
    int le = i / 48, c = i % 48;
    int e = e0 + le;
    float4 v = make_float4(0.f, 0.f, 0.f, 0.f);
    if (e < E) {
      if (c < 16)
        v = ((const float4*)(node_feat + (size_t)esrc[le] * DNODE))[c];
      else if (c < 32)
        v = ((const float4*)(node_feat + (size_t)edst[le] * DNODE))[c - 16];
      else
        v = ((const float4*)(edge_feat + (size_t)e * DNODE))[c - 32];
    }
    v4h h4 = {(_Float16)v.x, (_Float16)v.y, (_Float16)v.z, (_Float16)v.w};
    *(v4h*)&feat[le * DIN + c * 4] = h4;
  }
  __syncthreads();

  const int wave = tid >> 5, lane = tid & 31;
  const int mhi  = (lane & 16) ? 8 : 0;
  const int ncol = lane & 15;

  // ---- layer 1: K = 192 ----
  v8f acc[4] = {};
  const _Float16* arow = feat + wave * 16 * DIN;
#pragma unroll
  for (int kc = 0; kc < 6; ++kc) {
    v16h a = load_a16(arow, DIN, lane, kc);
#pragma unroll
    for (int nt = 0; nt < 4; ++nt) {
      v16h b = *(const v16h*)(wm1f + (kc * 4 + nt) * 512 + lane * 16);
      acc[nt] = __builtin_amdgcn_wmma_f32_16x16x32_f16(
          false, a, false, b, (short)0, acc[nt], false, false);
    }
  }
#pragma unroll
  for (int nt = 0; nt < 4; ++nt) {
    int n = nt * 16 + ncol;
    float bias = b_m1[n];
#pragma unroll
    for (int r = 0; r < 8; ++r) {
      float x = acc[nt][r] + bias;
      hbuf[(wave * 16 + mhi + r) * DNODE + n] = (_Float16)silu(x);
    }
  }

  // ---- layer 2: K = 64 ----
  v8f acc2[4] = {};
  const _Float16* hrow = hbuf + wave * 16 * DNODE;
#pragma unroll
  for (int kc = 0; kc < 2; ++kc) {
    v16h a = load_a16(hrow, DNODE, lane, kc);
#pragma unroll
    for (int nt = 0; nt < 4; ++nt) {
      v16h b = *(const v16h*)(wm2f + (kc * 4 + nt) * 512 + lane * 16);
      acc2[nt] = __builtin_amdgcn_wmma_f32_16x16x32_f16(
          false, a, false, b, (short)0, acc2[nt], false, false);
    }
  }

  // ---- scatter-add messages into agg[dst] ----
#pragma unroll
  for (int nt = 0; nt < 4; ++nt) {
    int n = nt * 16 + ncol;
    float bias = b_m2[n];
#pragma unroll
    for (int r = 0; r < 8; ++r) {
      int le = wave * 16 + mhi + r;
      int d  = edst[le];
      if (d >= 0) {
        float val = acc2[nt][r] + bias;
        unsafeAtomicAdd(&agg[(size_t)d * DNODE + n], val);
      }
    }
  }
}

// ---------------------------------------------------------------------------
// Node kernel: 64 nodes per block (4 waves x 16-node tiles).
//   upd = silu(agg @ Wu1 + b) @ Wu2 + b ; out = LN(node_feat + upd)
// ---------------------------------------------------------------------------
__global__ __launch_bounds__(128) void k_node(
    const float* __restrict__ node_feat,
    const float* __restrict__ agg,
    const _Float16* __restrict__ wu1f,
    const _Float16* __restrict__ wu2f,
    const float* __restrict__ b_u1,
    const float* __restrict__ b_u2,
    const float* __restrict__ gamma,
    const float* __restrict__ beta,
    float* __restrict__ out, int N) {
  __shared__ _Float16 abuf[64 * DNODE];
  __shared__ _Float16 hbuf[64 * DNODE];
  __shared__ float    ubuf[64 * DNODE];

  const int tid = threadIdx.x;
  const int n0  = blockIdx.x * 64;

  // load aggregated features -> f16 LDS (pad OOB rows with zero)
  for (int i = tid; i < 64 * 16; i += 128) {
    int r = i / 16, c = i % 16;
    int g = n0 + r;
    float4 v = make_float4(0.f, 0.f, 0.f, 0.f);
    if (g < N) v = ((const float4*)(agg + (size_t)g * DNODE))[c];
    v4h h4 = {(_Float16)v.x, (_Float16)v.y, (_Float16)v.z, (_Float16)v.w};
    *(v4h*)&abuf[r * DNODE + c * 4] = h4;
  }
  __syncthreads();

  const int wave = tid >> 5, lane = tid & 31;
  const int mhi  = (lane & 16) ? 8 : 0;
  const int ncol = lane & 15;

  // ---- update layer 1 ----
  v8f acc[4] = {};
  const _Float16* arow = abuf + wave * 16 * DNODE;
#pragma unroll
  for (int kc = 0; kc < 2; ++kc) {
    v16h a = load_a16(arow, DNODE, lane, kc);
#pragma unroll
    for (int nt = 0; nt < 4; ++nt) {
      v16h b = *(const v16h*)(wu1f + (kc * 4 + nt) * 512 + lane * 16);
      acc[nt] = __builtin_amdgcn_wmma_f32_16x16x32_f16(
          false, a, false, b, (short)0, acc[nt], false, false);
    }
  }
#pragma unroll
  for (int nt = 0; nt < 4; ++nt) {
    int n = nt * 16 + ncol;
    float bias = b_u1[n];
#pragma unroll
    for (int r = 0; r < 8; ++r) {
      float x = acc[nt][r] + bias;
      hbuf[(wave * 16 + mhi + r) * DNODE + n] = (_Float16)silu(x);
    }
  }

  // ---- update layer 2 ----
  v8f acc2[4] = {};
  const _Float16* hrow = hbuf + wave * 16 * DNODE;
#pragma unroll
  for (int kc = 0; kc < 2; ++kc) {
    v16h a = load_a16(hrow, DNODE, lane, kc);
#pragma unroll
    for (int nt = 0; nt < 4; ++nt) {
      v16h b = *(const v16h*)(wu2f + (kc * 4 + nt) * 512 + lane * 16);
      acc2[nt] = __builtin_amdgcn_wmma_f32_16x16x32_f16(
          false, a, false, b, (short)0, acc2[nt], false, false);
    }
  }
#pragma unroll
  for (int nt = 0; nt < 4; ++nt) {
    int n = nt * 16 + ncol;
    float bias = b_u2[n];
#pragma unroll
    for (int r = 0; r < 8; ++r)
      ubuf[(wave * 16 + mhi + r) * DNODE + n] = acc2[nt][r] + bias;
  }
  __syncthreads();

  // ---- residual + LayerNorm (one thread per node row) ----
  for (int r = tid; r < 64; r += 128) {
    int g = n0 + r;
    if (g >= N) continue;
    float sum = 0.f, sumsq = 0.f;
    for (int c = 0; c < DNODE; ++c) {
      float v = node_feat[(size_t)g * DNODE + c] + ubuf[r * DNODE + c];
      ubuf[r * DNODE + c] = v;   // stash residual-added value
      sum += v; sumsq += v * v;
    }
    float mu  = sum * (1.0f / DNODE);
    float var = sumsq * (1.0f / DNODE) - mu * mu;
    float inv = rsqrtf(var + 1e-5f);
    for (int c = 0; c < DNODE; ++c)
      out[(size_t)g * DNODE + c] =
          (ubuf[r * DNODE + c] - mu) * inv * gamma[c] + beta[c];
  }
}

// ---------------------------------------------------------------------------
extern "C" void kernel_launch(void* const* d_in, const int* in_sizes, int n_in,
                              void* d_out, int out_size, void* d_ws, size_t ws_size,
                              hipStream_t stream) {
  const float* node_feat = (const float*)d_in[0];
  const int*   edge_src  = (const int*)d_in[1];
  const int*   edge_dst  = (const int*)d_in[2];
  const float* edge_feat = (const float*)d_in[3];
  const float* W_m1 = (const float*)d_in[4];
  const float* b_m1 = (const float*)d_in[5];
  const float* W_m2 = (const float*)d_in[6];
  const float* b_m2 = (const float*)d_in[7];
  const float* W_u1 = (const float*)d_in[8];
  const float* b_u1 = (const float*)d_in[9];
  const float* W_u2 = (const float*)d_in[10];
  const float* b_u2 = (const float*)d_in[11];
  const float* ln_g = (const float*)d_in[12];
  const float* ln_b = (const float*)d_in[13];

  const int N = in_sizes[0] / DNODE;   // nodes
  const int E = in_sizes[1];           // edges

  // workspace: [agg f32 N*64 | weight fragments f16]
  float* agg = (float*)d_ws;
  size_t aggBytes = (((size_t)N * DNODE * sizeof(float)) + 255) & ~(size_t)255;
  _Float16* frag = (_Float16*)((char*)d_ws + aggBytes);
  _Float16* wm1f = frag;
  _Float16* wm2f = frag + 24 * 512;
  _Float16* wu1f = frag + 32 * 512;
  _Float16* wu2f = frag + 40 * 512;

  k_zero<<<(N * DNODE + 255) / 256, 256, 0, stream>>>(agg, N * DNODE);
  k_prep<<<(48 * 512 + 255) / 256, 256, 0, stream>>>(W_m1, W_m2, W_u1, W_u2, frag);
  k_edge<<<(E + 63) / 64, 128, 0, stream>>>(node_feat, edge_src, edge_dst, edge_feat,
                                            wm1f, wm2f, b_m1, b_m2, agg, E);
  k_node<<<(N + 63) / 64, 128, 0, stream>>>(node_feat, agg, wu1f, wu2f,
                                            b_u1, b_u2, ln_g, ln_b,
                                            (float*)d_out, N);
}
